// RQAutoEncoder_8753143349509
// MI455X (gfx1250) — matrix operand order repaired and checked by reading, not compiled, not run on hardware
//
#include <hip/hip_runtime.h>
#include <hip/hip_bf16.h>

// ---------------------------------------------------------------------------
// RQ-VAE forward for MI455X (gfx1250, wave32, WMMA + async LDS copies).
//   h  = relu(x @ enc_w1 + b1)          [32768,512]   bf16 WMMA (A fp32->bf16)
//   z  = h @ enc_w2 + b2                [32768,32]    bf16 WMMA (A async->LDS)
//   zq, idx, loss = residual VQ(z)                    wave-parallel
//   d  = relu(zq @ dec_w1 + b1)         [32768,512]   bf16 WMMA (A async->LDS)
//   out= d @ dec_w2 + b2                [32768,4096]  bf16 WMMA (A async->LDS)
// ---------------------------------------------------------------------------

typedef __bf16 bf16_t;
typedef __attribute__((ext_vector_type(16))) __bf16        v16bf;
typedef __attribute__((ext_vector_type(8)))  float         v8f;
typedef __attribute__((ext_vector_type(8)))  unsigned int  v8u;

#define D_IN    4096
#define D_HID   512
#define D_CODE  32
#define NUM_Q   3
#define CB_SIZE 256
#define NROWS   32768

// GEMM tiling: block = 256 threads = 8 waves; tile 128(M) x TNV(N), K step 32.
#define TM 128
#define TK 32
#define LA_LD 40   // padded LDS row (bf16): 80B stride = 5 x 16B (b128-aligned),
                   // 20 DWORDs -> co-prime with 64 banks for the gather phase

// gfx1250 async global->LDS copy (ASYNCcnt-tracked), GV addressing mode.
__device__ __forceinline__ void async_copy_b128(unsigned int ldsAddr,
                                                const void* gptr) {
    asm volatile("global_load_async_to_lds_b128 %0, %1, off"
                 :: "v"(ldsAddr), "v"((unsigned long long)(uintptr_t)gptr)
                 : "memory");
}
__device__ __forceinline__ void wait_async0() {
    asm volatile("s_wait_asynccnt 0" ::: "memory");
}

// ---------------------------------------------------------------------------
// Generic bf16-WMMA GEMM:  C[M,Ntot] = act(A[M,Ktot] * B[Ktot,Ntot] + bias)
//   A fp32: register-staged + v_cvt_pk_bf16_f32.  A bf16: async copy to LDS.
//   B fp32: register-staged + converted, stored transposed [n][k].
//   TNV in {32, 64} chosen so all tilings divide exactly (no guards).
//   Double-buffered LDS software pipeline.
// ---------------------------------------------------------------------------
template <int TNV, bool A_IS_F32, bool RELU, bool OUT_BF16>
__global__ __launch_bounds__(256) void rq_gemm_wmma(
    const void* __restrict__ Aptr, const float* __restrict__ B,
    const float* __restrict__ bias, void* __restrict__ Cptr,
    int Ktot, int Ntot)
{
    constexpr int NSUB  = TNV / 16;   // 16x16 accumulators per wave
    constexpr int BITER = TNV / 32;   // float4 B-loads per thread
    constexpr int BCOLS = TNV / 4;    // float4 slots per B k-row

    __shared__ bf16_t la[2][TM * LA_LD];    // A tile, [m][k]
    __shared__ bf16_t lb[2][TNV * LA_LD];   // B tile transposed, [n][k]

    const int tid  = threadIdx.x;
    const int lane = tid & 31;
    const int wave = tid >> 5;           // 0..7 -> M subtile
    const int half = (lane >> 4) & 1;    // K/M-half select per ISA layout
    const int l16  = lane & 15;

    const int mBase = blockIdx.y * TM;
    const int nBase = blockIdx.x * TNV;

    // register staging for the in-flight tile (fp32 paths only)
    float4 aF[4];
    float4 bR[BITER];

    // issue A-tile staging for k0 into LDS buffer `buf`
    auto loadA = [&](int k0, int buf) {
        if constexpr (A_IS_F32) {
            const float* A = (const float*)Aptr;
#pragma unroll
            for (int i = 0; i < 4; ++i) {
                const int lin = tid + 256 * i;     // 1024 float4 slots
                const int row = lin >> 3;          // 8 float4 per 32-elem row
                const int kk  = (lin & 7) * 4;
                aF[i] = *(const float4*)(A + (size_t)(mBase + row) * Ktot + k0 + kk);
            }
            (void)buf;
        } else {
            const bf16_t* A = (const bf16_t*)Aptr;
#pragma unroll
            for (int i = 0; i < 2; ++i) {
                const int lin = tid + 256 * i;     // 512 b128 slots (8 bf16)
                const int row = lin >> 2;          // 4 b128 per 32-elem row
                const int kk  = (lin & 3) * 8;
                const unsigned int ldsAddr =
                    (unsigned int)(uintptr_t)&la[buf][row * LA_LD + kk];
                async_copy_b128(ldsAddr,
                                A + (size_t)(mBase + row) * Ktot + k0 + kk);
            }
        }
    };

    // fp32-A conversion + LDS store (no-op for bf16 path; async already wrote)
    auto storeA = [&](int buf) {
        if constexpr (A_IS_F32) {
#pragma unroll
            for (int i = 0; i < 4; ++i) {
                const int lin = tid + 256 * i;
                const int row = lin >> 3;
                const int kk  = (lin & 7) * 4;
                bf16_t* dst = &la[buf][row * LA_LD + kk];
                dst[0] = (bf16_t)aF[i].x; dst[1] = (bf16_t)aF[i].y;
                dst[2] = (bf16_t)aF[i].z; dst[3] = (bf16_t)aF[i].w;
            }
        }
    };

    auto loadB = [&](int k0) {
#pragma unroll
        for (int i = 0; i < BITER; ++i) {
            const int lin  = tid + 256 * i;        // TK*TNV/4 float4 slots
            const int krow = lin / BCOLS;
            const int nn   = (lin % BCOLS) * 4;
            bR[i] = *(const float4*)(B + (size_t)(k0 + krow) * Ntot + nBase + nn);
        }
    };

    auto storeB = [&](int buf) {
#pragma unroll
        for (int i = 0; i < BITER; ++i) {
            const int lin  = tid + 256 * i;
            const int krow = lin / BCOLS;
            const int nn   = (lin % BCOLS) * 4;
            lb[buf][(nn + 0) * LA_LD + krow] = (bf16_t)bR[i].x;
            lb[buf][(nn + 1) * LA_LD + krow] = (bf16_t)bR[i].y;
            lb[buf][(nn + 2) * LA_LD + krow] = (bf16_t)bR[i].z;
            lb[buf][(nn + 3) * LA_LD + krow] = (bf16_t)bR[i].w;
        }
    };

    v8f acc[NSUB];
#pragma unroll
    for (int i = 0; i < NSUB; ++i) acc[i] = v8f{};

    const int numK = Ktot / TK;

    // prologue: stage tile 0 into buffer 0
    loadA(0, 0);
    loadB(0);
    storeA(0);
    storeB(0);
    if constexpr (!A_IS_F32) wait_async0();   // async LDS writes must land
                                              // before the first barrier

    int buf = 0;
    for (int t = 0; t < numK; ++t) {
        __syncthreads();
        if (t + 1 < numK) {                   // overlap tile t+1 with WMMA on t
            loadA((t + 1) * TK, buf ^ 1);     // bf16: async straight into LDS
            loadB((t + 1) * TK);
        }
        if (t + 2 < numK) {                   // GL2 prefetch two tiles ahead
            const char*  base = (const char*)Aptr;
            const size_t esz  = A_IS_F32 ? 4 : 2;
            __builtin_prefetch(
                base + ((size_t)(mBase + (tid >> 1)) * Ktot + (t + 2) * TK) * esz, 0, 1);
        }

        // ---- A fragment (16x32 bf16, k = 16*half + s, same order as B) ----
        const unsigned int* laRow =
            (const unsigned int*)&la[buf][(wave * 16 + l16) * LA_LD];
        v8u au;
#pragma unroll
        for (int p = 0; p < 8; ++p) {
            const int j = 2 * p;
            const int k = half * 16 + j;
            au[p] = laRow[k >> 1];
        }
        const v16bf afrag = __builtin_bit_cast(v16bf, au);

        // ---- NSUB B fragments + WMMA ----
#pragma unroll
        for (int nt = 0; nt < NSUB; ++nt) {
            const unsigned int* lbRow =
                (const unsigned int*)&lb[buf][(nt * 16 + l16) * LA_LD];
            v8u bu;
#pragma unroll
            for (int p = 0; p < 8; ++p) bu[p] = lbRow[half * 8 + p];
            const v16bf bfrag = __builtin_bit_cast(v16bf, bu);
            acc[nt] = __builtin_amdgcn_wmma_f32_16x16x32_bf16(
                false, afrag, false, bfrag, (short)0, acc[nt], false, false);
        }

        if (t + 1 < numK) {
            storeA(buf ^ 1);                  // fp32 path converts+stores here
            storeB(buf ^ 1);
        }
        if constexpr (!A_IS_F32) wait_async0();  // drain before next barrier
        buf ^= 1;
    }

    // ---- epilogue: bias (+relu), store per C/D VGPR layout ----
#pragma unroll
    for (int nt = 0; nt < NSUB; ++nt) {
        const int   n  = nBase + nt * 16 + l16;
        const float bv = bias[n];
#pragma unroll
        for (int r = 0; r < 8; ++r) {
            const int m = mBase + wave * 16 + half * 8 + r;
            float v = acc[nt][r] + bv;
            if (RELU) v = v > 0.f ? v : 0.f;
            if (OUT_BF16)
                ((bf16_t*)Cptr)[(size_t)m * Ntot + n] = (bf16_t)v;
            else
                ((float*)Cptr)[(size_t)m * Ntot + n] = v;
        }
    }
}

// ---------------------------------------------------------------------------
// Residual VQ: one row per wave (8 rows / block). Codebook + ||c||^2 in LDS.
// dist = ||r||^2 - 2 r.c + ||c||^2 ; argmin only needs ||c||^2 - 2 r.c.
// ---------------------------------------------------------------------------
__global__ __launch_bounds__(256) void rq_rvq_kernel(
    const float* __restrict__ z, const float* __restrict__ codebooks,
    bf16_t* __restrict__ zq, int* __restrict__ idxOut,
    float* __restrict__ lossPartial)
{
    __shared__ float cb[CB_SIZE * D_CODE];   // 32 KB
    __shared__ float cnorm[CB_SIZE];
    __shared__ float lossW[NUM_Q][8];

    const int tid  = threadIdx.x;
    const int lane = tid & 31;
    const int wave = tid >> 5;
    const int row  = blockIdx.x * 8 + wave;

    float r[D_CODE], zacc[D_CODE];
#pragma unroll
    for (int i = 0; i < D_CODE; ++i) {
        r[i]    = z[(size_t)row * D_CODE + i];
        zacc[i] = 0.f;
    }

    for (int q = 0; q < NUM_Q; ++q) {
        __syncthreads();
        for (int i = tid; i < CB_SIZE * D_CODE; i += 256)
            cb[i] = codebooks[(size_t)q * CB_SIZE * D_CODE + i];
        __syncthreads();
        if (tid < CB_SIZE) {
            float s = 0.f;
#pragma unroll
            for (int i = 0; i < D_CODE; ++i) {
                const float c = cb[tid * D_CODE + i];
                s += c * c;
            }
            cnorm[tid] = s;
        }
        __syncthreads();

        float bestD = 3.4e38f;
        int   bestI = 0;
        for (int t = 0; t < 8; ++t) {
            const int c = lane + 32 * t;     // per-lane ascending -> tie-break ok
            float dot = 0.f;
#pragma unroll
            for (int i = 0; i < D_CODE; ++i) dot += r[i] * cb[c * D_CODE + i];
            const float d = cnorm[c] - 2.f * dot;
            if (d < bestD || (d == bestD && c < bestI)) { bestD = d; bestI = c; }
        }
        // wave32 butterfly argmin (tie -> smallest index, matches jnp.argmin)
#pragma unroll
        for (int off = 16; off > 0; off >>= 1) {
            const float od = __shfl_xor(bestD, off, 32);
            const int   oi = __shfl_xor(bestI, off, 32);
            if (od < bestD || (od == bestD && oi < bestI)) { bestD = od; bestI = oi; }
        }
        const int idx = bestI;

        // commitment loss contribution: lane l owns dim l
        const float dl = cb[idx * D_CODE + lane] - r[lane];
        float ls = dl * dl;
#pragma unroll
        for (int off = 16; off > 0; off >>= 1) ls += __shfl_xor(ls, off, 32);
        if (lane == 0) {
            lossW[q][wave] = ls;
            idxOut[(size_t)row * NUM_Q + q] = idx;
        }
#pragma unroll
        for (int i = 0; i < D_CODE; ++i) {
            const float qv = cb[idx * D_CODE + i];
            zacc[i] += qv;          // straight-through forward value == quant
            r[i]    -= qv;
        }
    }
    zq[(size_t)row * D_CODE + lane] = (bf16_t)zacc[lane];

    __syncthreads();
    if (tid == 0) {
        for (int q = 0; q < NUM_Q; ++q) {
            float s = 0.f;
            for (int w = 0; w < 8; ++w) s += lossW[q][w];
            lossPartial[(size_t)blockIdx.x * NUM_Q + q] = s;
        }
    }
}

// Deterministic single-block final loss reduction.
__global__ __launch_bounds__(256) void rq_loss_finalize(
    const float* __restrict__ partial, float* __restrict__ out, int nBlocks)
{
    __shared__ float red[256];
    const int tid = threadIdx.x;
    for (int q = 0; q < NUM_Q; ++q) {
        float s = 0.f;
        for (int b = tid; b < nBlocks; b += 256) s += partial[(size_t)b * NUM_Q + q];
        red[tid] = s;
        __syncthreads();
        for (int off = 128; off > 0; off >>= 1) {
            if (tid < off) red[tid] += red[tid + off];
            __syncthreads();
        }
        if (tid == 0) out[q] = red[0] / (float)((size_t)NROWS * D_CODE);
        __syncthreads();
    }
}

// ---------------------------------------------------------------------------
extern "C" void kernel_launch(void* const* d_in, const int* in_sizes, int n_in,
                              void* d_out, int out_size, void* d_ws, size_t ws_size,
                              hipStream_t stream) {
    (void)in_sizes; (void)n_in; (void)out_size; (void)ws_size;

    const float* x         = (const float*)d_in[0];
    const float* enc_w1    = (const float*)d_in[1];
    const float* enc_b1    = (const float*)d_in[2];
    const float* enc_w2    = (const float*)d_in[3];
    const float* enc_b2    = (const float*)d_in[4];
    const float* dec_w1    = (const float*)d_in[5];
    const float* dec_b1    = (const float*)d_in[6];
    const float* dec_w2    = (const float*)d_in[7];
    const float* dec_b2    = (const float*)d_in[8];
    const float* codebooks = (const float*)d_in[9];

    // outputs: out[N,4096] f32 | indices[N,3] i32 | commit_loss[3] f32
    float* outF    = (float*)d_out;
    int*   idxOut  = (int*)(outF + (size_t)NROWS * D_IN);
    float* lossOut = outF + (size_t)NROWS * D_IN + (size_t)NROWS * NUM_Q;

    // workspace carve-up
    char* ws = (char*)d_ws;
    bf16_t* h_bf16      = (bf16_t*)ws;                       ws += (size_t)NROWS * D_HID  * sizeof(bf16_t);
    float*  z_f32       = (float*)ws;                        ws += (size_t)NROWS * D_CODE * sizeof(float);
    bf16_t* zq_bf16     = (bf16_t*)ws;                       ws += (size_t)NROWS * D_CODE * sizeof(bf16_t);
    bf16_t* d_bf16      = (bf16_t*)ws;                       ws += (size_t)NROWS * D_HID  * sizeof(bf16_t);
    float*  lossPartial = (float*)ws;

    const dim3 blk(256);
    const int  mBlocks = NROWS / TM;       // 256

    // h = relu(x @ enc_w1 + b1)
    rq_gemm_wmma<64, true, true, true><<<dim3(D_HID / 64, mBlocks), blk, 0, stream>>>(
        x, enc_w1, enc_b1, h_bf16, D_IN, D_HID);
    // z = h @ enc_w2 + b2   (exact 32-wide tile, async A staging)
    rq_gemm_wmma<32, false, false, false><<<dim3(1, mBlocks), blk, 0, stream>>>(
        h_bf16, enc_w2, enc_b2, z_f32, D_HID, D_CODE);
    // residual VQ
    rq_rvq_kernel<<<dim3(NROWS / 8), blk, 0, stream>>>(
        z_f32, codebooks, zq_bf16, idxOut, lossPartial);
    // d = relu(zq @ dec_w1 + b1)
    rq_gemm_wmma<64, false, true, true><<<dim3(D_HID / 64, mBlocks), blk, 0, stream>>>(
        zq_bf16, dec_w1, dec_b1, d_bf16, D_CODE, D_HID);
    // out = d @ dec_w2 + b2
    rq_gemm_wmma<64, false, false, false><<<dim3(D_IN / 64, mBlocks), blk, 0, stream>>>(
        d_bf16, dec_w2, dec_b2, outF, D_HID, D_IN);
    // loss
    rq_loss_finalize<<<dim3(1), blk, 0, stream>>>(lossPartial, lossOut, NROWS / 8);
}
